// GATEncoder_58256936403625
// MI455X (gfx1250) — matrix operand order repaired and checked by reading, not compiled, not run on hardware
//
#include <hip/hip_runtime.h>
#include <hip/hip_bf16.h>

// ---------------------------------------------------------------------------
// GAT encoder for MI455X (gfx1250, wave32).
// GEMMs: bf16 WMMA (v_wmma_f32_16x16x32_bf16) with f32 accumulation.
//   - W is transpose-converted to bf16 once per layer, so both A and B tiles
//     stage to LDS with a single b128 global load + b128 ds store per thread.
// Graph phase: L2-resident gather/scatter with f32 atomics (memory bound).
// ---------------------------------------------------------------------------

typedef __attribute__((ext_vector_type(16))) __bf16 v16bf;
typedef __attribute__((ext_vector_type(8)))  float  v8f;

#define KDIM 256      // inner dim is 256 for all three layers
#define BM 64
#define BN 64
#define BK 32
#define ASTR 40       // 32 + 8 bf16 padding -> 80B row stride, decorrelates banks

__device__ __forceinline__ unsigned short f2bf(float f) {
    unsigned u = __float_as_uint(f);
    unsigned r = (u + 0x7FFFu + ((u >> 16) & 1u)) >> 16;   // RNE
    return (unsigned short)r;
}

// ---------------- fp32 -> bf16 conversion (grid-stride) --------------------
__global__ void k_f32_to_bf16(const float* __restrict__ in,
                              unsigned short* __restrict__ out, long long n) {
    long long i = (long long)blockIdx.x * blockDim.x + threadIdx.x;
    long long stride = (long long)gridDim.x * blockDim.x;
    for (; i < n; i += stride) out[i] = f2bf(in[i]);
}

// ------- fp32 W[K][Nout] -> bf16 Wt[Nout][K] (transpose during convert) -----
__global__ void k_w_to_bf16_T(const float* __restrict__ W,
                              unsigned short* __restrict__ Wt,
                              int K, int Nout) {
    long long idx = (long long)blockIdx.x * blockDim.x + threadIdx.x;
    if (idx >= (long long)K * Nout) return;
    int k = (int)(idx / Nout), n = (int)(idx % Nout);
    Wt[(long long)n * K + k] = f2bf(W[idx]);
}

// ---------------- zero fill (floats / uint keys share bit pattern) ----------
__global__ void k_zero_f32(float* __restrict__ p, long long n) {
    long long i = (long long)blockIdx.x * blockDim.x + threadIdx.x;
    long long stride = (long long)gridDim.x * blockDim.x;
    for (; i < n; i += stride) p[i] = 0.0f;
}

// ---------------- bf16 WMMA GEMM: C[M,Nout] = A[M,256] * Wt[Nout,256]^T -----
__global__ __launch_bounds__(256)
void k_gemm_bf16_wmma(const unsigned short* __restrict__ A,    // [M,KDIM] bf16
                      const unsigned short* __restrict__ Wt,   // [Nout,KDIM] bf16
                      float* __restrict__ C, int M, int Nout) {
    __shared__ unsigned short As[BM * ASTR];   // [m][k]
    __shared__ unsigned short Bs[BN * ASTR];   // [n][k]  (B already transposed)

    const int t    = threadIdx.x;
    const int lane = t & 31;
    const int w    = t >> 5;          // 8 waves
    const int wm   = w >> 2;          // 2 along M (32 rows each)
    const int wn   = w & 3;           // 4 along N (16 cols each)
    const int half = lane >> 4;       // lane half selects K sub-range
    const int ml   = lane & 15;

    const int m0 = blockIdx.x * BM;
    const int n0 = blockIdx.y * BN;

    // staging coordinates (same pattern for A and B): row = t>>2, 8-elem chunk
    const int sr = t >> 2;
    const int sk = (t & 3) * 8;
    const bool aInR = (m0 + sr) < M;
    const bool bInR = (n0 + sr) < Nout;        // Nout is a multiple of 8
    const unsigned short* aSrc = A  + (long long)(m0 + sr) * KDIM + sk;
    const unsigned short* bSrc = Wt + (long long)(n0 + sr) * KDIM + sk;

    v8f c0 = {};   // rows [wm*32 .. +16)
    v8f c1 = {};   // rows [wm*32+16 .. +32)

    for (int k0 = 0; k0 < KDIM; k0 += BK) {
        // -- stage A and B tiles: one guarded b128 load + b128 LDS store each
        {
            uint4 va = {0u, 0u, 0u, 0u};
            uint4 vb = {0u, 0u, 0u, 0u};
            if (aInR) va = *(const uint4*)(aSrc + k0);
            if (bInR) vb = *(const uint4*)(bSrc + k0);
            *(uint4*)(&As[sr * ASTR + sk]) = va;
            *(uint4*)(&Bs[sr * ASTR + sk]) = vb;
        }
        // prefetch next K-step source lines (global_prefetch_b8)
        if (k0 + BK < KDIM) {
            __builtin_prefetch(aSrc + k0 + BK, 0, 3);
            __builtin_prefetch(bSrc + k0 + BK, 0, 3);
        }
        __syncthreads();

        // -- A fragments: lane half h holds K in [8h,8h+8) and [8h+16,8h+24)
        union Frag { v16bf v; uint4 q[2]; } a0, a1, bf;
        const int kb = half * 8;
        {
            const unsigned short* ap = &As[(wm * 32 + ml) * ASTR];
            a0.q[0] = *(const uint4*)(ap + kb);
            a0.q[1] = *(const uint4*)(ap + kb + 16);
            const unsigned short* ap2 = ap + 16 * ASTR;
            a1.q[0] = *(const uint4*)(ap2 + kb);
            a1.q[1] = *(const uint4*)(ap2 + kb + 16);
        }
        // -- B fragment: lane = col (ml), half h holds K in [8h,8h+8) and [8h+16,8h+24)
        {
            const unsigned short* bp = &Bs[(wn * 16 + ml) * ASTR + half * 8];
            bf.q[0] = *(const uint4*)(bp);
            bf.q[1] = *(const uint4*)(bp + 16);
        }

        c0 = __builtin_amdgcn_wmma_f32_16x16x32_bf16(
                 false, a0.v, false, bf.v, (short)0, c0, false, false);
        c1 = __builtin_amdgcn_wmma_f32_16x16x32_bf16(
                 false, a1.v, false, bf.v, (short)0, c1, false, false);
        __syncthreads();
    }

    // -- store D: VGPR r -> row r + 8*half, col = ml
    const int colg = n0 + wn * 16 + ml;
    if (colg < Nout) {
#pragma unroll
        for (int r = 0; r < 8; ++r) {
            int row0 = m0 + wm * 32 + half * 8 + r;
            if (row0 < M)      C[(long long)row0 * Nout + colg] = c0[r];
            int row1 = row0 + 16;
            if (row1 < M)      C[(long long)row1 * Nout + colg] = c1[r];
        }
    }
}

// ---------------- per-node attention scores el/er ---------------------------
__global__ void k_node_scores(const float* __restrict__ feat,
                              const float* __restrict__ al,
                              const float* __restrict__ ar,
                              float* __restrict__ el, float* __restrict__ er,
                              int N, int H, int D) {
    long long idx = (long long)blockIdx.x * blockDim.x + threadIdx.x;
    if (idx >= (long long)N * H) return;
    int n = (int)(idx / H), h = (int)(idx % H);
    const float* f = feat + (long long)n * H * D + h * D;
    float sl = 0.f, sr = 0.f;
    for (int d = 0; d < D; ++d) {
        float v = f[d];
        sl += v * al[h * D + d];
        sr += v * ar[h * D + d];
    }
    el[idx] = sl;
    er[idx] = sr;
}

__device__ __forceinline__ float leaky(float v) { return v > 0.f ? v : 0.2f * v; }
__device__ __forceinline__ unsigned fkey(float v) {
    unsigned u = __float_as_uint(v);
    return (u >> 31) ? ~u : (u | 0x80000000u);
}
__device__ __forceinline__ float fdec(unsigned k) {
    return (k >> 31) ? __uint_as_float(k & 0x7FFFFFFFu) : __uint_as_float(~k);
}

// ---------------- segment max over dst (float key atomic max) ---------------
__global__ void k_edge_max(const float* __restrict__ el, const float* __restrict__ er,
                           const int* __restrict__ src, const int* __restrict__ dst,
                           unsigned* __restrict__ emaxk, long long EH, int H) {
    long long idx = (long long)blockIdx.x * blockDim.x + threadIdx.x;
    if (idx >= EH) return;
    int e = (int)(idx / H), h = (int)(idx % H);
    float v = leaky(el[(long long)src[e] * H + h] + er[(long long)dst[e] * H + h]);
    atomicMax(&emaxk[(long long)dst[e] * H + h], fkey(v));
}

// ---------------- exp(e - max) and segment sum ------------------------------
__global__ void k_edge_expsum(const float* __restrict__ el, const float* __restrict__ er,
                              const int* __restrict__ src, const int* __restrict__ dst,
                              const unsigned* __restrict__ emaxk,
                              float* __restrict__ denom, float* __restrict__ ex,
                              long long EH, int H) {
    long long idx = (long long)blockIdx.x * blockDim.x + threadIdx.x;
    if (idx >= EH) return;
    int e = (int)(idx / H), h = (int)(idx % H);
    int d = dst[e];
    float v = leaky(el[(long long)src[e] * H + h] + er[(long long)d * H + h]);
    float m = fdec(emaxk[(long long)d * H + h]);
    float x = __expf(v - m);
    ex[idx] = x;
    atomicAdd(&denom[(long long)d * H + h], x);
}

// ---------------- weighted scatter aggregation ------------------------------
__global__ void k_aggregate(const float* __restrict__ feat,
                            const float* __restrict__ ex, const float* __restrict__ denom,
                            const int* __restrict__ src, const int* __restrict__ dst,
                            float* __restrict__ agg, long long EH, int H, int D) {
    long long idx = (long long)blockIdx.x * blockDim.x + threadIdx.x;
    if (idx >= EH) return;
    int e = (int)(idx / H), h = (int)(idx % H);
    int s = src[e], d = dst[e];
    float a = ex[idx] / (denom[(long long)d * H + h] + 1e-9f);
    const float* f = feat + (long long)s * H * D + h * D;
    float*       o = agg  + (long long)d * H * D + h * D;
#pragma unroll 8
    for (int j = 0; j < D; ++j) atomicAdd(&o[j], f[j] * a);
}

// ---------------- bias + ELU (in place) -------------------------------------
__global__ void k_bias_elu(float* __restrict__ agg, const float* __restrict__ b,
                           long long total, int HD) {
    long long idx = (long long)blockIdx.x * blockDim.x + threadIdx.x;
    if (idx >= total) return;
    float v = agg[idx] + b[idx % HD];
    agg[idx] = v > 0.f ? v : (expm1f(v));
}

// ---------------- final bias (head mean of single head == identity) ---------
__global__ void k_bias_final(const float* __restrict__ agg, const float* __restrict__ b,
                             float* __restrict__ out, long long total, int HD) {
    long long idx = (long long)blockIdx.x * blockDim.x + threadIdx.x;
    if (idx >= total) return;
    out[idx] = agg[idx] + b[idx % HD];
}

// ---------------------------------------------------------------------------
static inline int gridFor(long long n, int b) { return (int)((n + b - 1) / b); }

extern "C" void kernel_launch(void* const* d_in, const int* in_sizes, int n_in,
                              void* d_out, int out_size, void* d_ws, size_t ws_size,
                              hipStream_t stream) {
    const float* features = (const float*)d_in[0];
    const int*   src      = (const int*)d_in[1];
    const int*   dst      = (const int*)d_in[2];
    const float* W[3]  = {(const float*)d_in[3],  (const float*)d_in[7],  (const float*)d_in[11]};
    const float* al[3] = {(const float*)d_in[4],  (const float*)d_in[8],  (const float*)d_in[12]};
    const float* ar[3] = {(const float*)d_in[5],  (const float*)d_in[9],  (const float*)d_in[13]};
    const float* bb[3] = {(const float*)d_in[6],  (const float*)d_in[10], (const float*)d_in[14]};

    const int N = in_sizes[0] / KDIM;       // 50000
    const int E = in_sizes[1];              // 800000
    const int HMAX = 8, DD = 32;

    // ---- workspace carve-up (256B aligned) ----
    size_t off = 0;
    auto carve = [&](size_t bytes) -> char* {
        char* p = (char*)d_ws + off;
        off += (bytes + 255) & ~(size_t)255;
        return p;
    };
    unsigned short* xb   = (unsigned short*)carve((size_t)N * KDIM * 2);
    unsigned short* wtb  = (unsigned short*)carve((size_t)KDIM * KDIM * 2);
    float*          feat = (float*)carve((size_t)N * KDIM * 4);
    float*          el   = (float*)carve((size_t)N * HMAX * 4);
    float*          er   = (float*)carve((size_t)N * HMAX * 4);
    unsigned*       emk  = (unsigned*)carve((size_t)N * HMAX * 4);
    float*          den  = (float*)carve((size_t)N * HMAX * 4);
    float*          exb  = (float*)carve((size_t)E * HMAX * 4);
    float*          agg  = (float*)carve((size_t)N * KDIM * 4);
    (void)ws_size;

    const int TB = 256;
    const float* x = features;

    for (int l = 0; l < 3; ++l) {
        const int H    = (l == 2) ? 1 : 8;
        const int Nout = H * DD;
        const long long NH  = (long long)N * H;
        const long long EH  = (long long)E * H;
        const long long NNo = (long long)N * Nout;

        // 1. fp32 -> bf16: activations (straight), weights (transposed)
        k_f32_to_bf16<<<1024, TB, 0, stream>>>(x, xb, (long long)N * KDIM);
        k_w_to_bf16_T<<<gridFor((long long)KDIM * Nout, TB), TB, 0, stream>>>(W[l], wtb, KDIM, Nout);

        // 2. feat = x @ W  (bf16 WMMA, f32 accumulate)
        dim3 gg(gridFor(N, BM), gridFor(Nout, BN));
        k_gemm_bf16_wmma<<<gg, TB, 0, stream>>>(xb, wtb, feat, N, Nout);

        // 3. attention scores
        k_node_scores<<<gridFor(NH, TB), TB, 0, stream>>>(feat, al[l], ar[l], el, er, N, H, DD);

        // 4. init segment buffers (zero bits == -inf key for max)
        k_zero_f32<<<gridFor(NH, TB), TB, 0, stream>>>((float*)emk, NH);
        k_zero_f32<<<gridFor(NH, TB), TB, 0, stream>>>(den, NH);
        k_zero_f32<<<gridFor(NNo, TB), TB, 0, stream>>>(agg, NNo);

        // 5. segment softmax + weighted scatter aggregation
        k_edge_max<<<gridFor(EH, TB), TB, 0, stream>>>(el, er, src, dst, emk, EH, H);
        k_edge_expsum<<<gridFor(EH, TB), TB, 0, stream>>>(el, er, src, dst, emk, den, exb, EH, H);
        k_aggregate<<<gridFor(EH, TB), TB, 0, stream>>>(feat, exb, den, src, dst, agg, EH, H, DD);

        // 6. bias + activation
        if (l < 2) {
            k_bias_elu<<<gridFor(NNo, TB), TB, 0, stream>>>(agg, bb[l], NNo, Nout);
            x = agg;     // next layer input (re-converted to bf16 before reuse)
        } else {
            k_bias_final<<<gridFor(NNo, TB), TB, 0, stream>>>(agg, bb[l], (float*)d_out, NNo, Nout);
        }
    }
}
